// Head_27745488732305
// MI455X (gfx1250) — matrix operand order repaired and checked by reading, not compiled, not run on hardware
//
#include <hip/hip_runtime.h>
#include <hip/hip_bf16.h>
#include <cstddef>

// ---------------------------------------------------------------------------
// Single attention head, bf16 WMMA flash-attention for gfx1250 (wave32),
// async global->LDS double-buffered staging (ASYNCcnt path); all fragments
// of a step are loaded into distinct registers before the WMMA chain so the
// waitcnt pass can overlap LDS latency with matrix math.
//   B=16, T=2048, C=768, H=64
// ---------------------------------------------------------------------------

typedef __attribute__((ext_vector_type(16))) __bf16 v16bf;
typedef __attribute__((ext_vector_type(8)))  float  v8f;

constexpr int kB = 16;
constexpr int kT = 2048;
constexpr int kC = 768;
constexpr int kH = 64;
constexpr float kScale = 0.125f;         // 1/sqrt(64)
constexpr float kNegInf = -3.0e38f;

// Workspace layout (bytes)
constexpr size_t kWtBytes  = (size_t)3 * kH * kC * 2;        // 294912
constexpr size_t kMatBytes = (size_t)kB * kT * kH * 2;       // 4 MiB each
constexpr size_t kOffK  = kWtBytes;
constexpr size_t kOffQ  = kWtBytes + kMatBytes;
constexpr size_t kOffVt = kWtBytes + 2 * kMatBytes;

__device__ inline void cvt4(__bf16* d, float4 f) {
    d[0] = (__bf16)f.x; d[1] = (__bf16)f.y; d[2] = (__bf16)f.z; d[3] = (__bf16)f.w;
}

union Frag {            // one WMMA 16-bit operand: 16 bf16 per lane (8 VGPRs)
    v16bf  v;
    float4 q[2];
    __bf16 e[16];
};

__device__ inline v8f wmma_bf16(const Frag& a, const Frag& b, v8f c) {
    return __builtin_amdgcn_wmma_f32_16x16x32_bf16(
        false, a.v, false, b.v, (short)0, c, false, false);
}

// --- CDNA5 async global->LDS copy (16 B per lane), tracked by ASYNCcnt ---
__device__ inline void async_load_b128(unsigned lds_off, const void* gaddr) {
    asm volatile("global_load_async_to_lds_b128 %0, %1, off"
                 :: "v"(lds_off), "v"(gaddr)
                 : "memory");
}
__device__ inline void wait_async() {
#if __has_builtin(__builtin_amdgcn_s_wait_asynccnt)
    __builtin_amdgcn_s_wait_asynccnt(0);
#else
    asm volatile("s_wait_asynccnt 0x0" ::: "memory");
#endif
}
__device__ inline unsigned lds_offset(const void* p) {
    return (unsigned)(unsigned long long)p;   // low 32 bits of generic = LDS offset
}

// ---------------------------------------------------------------------------
// Kernel 0: W[c][h] f32 -> Wt[h][c] bf16 for Wk, Wq, Wv
// ---------------------------------------------------------------------------
__global__ void wt_convert(const float* __restrict__ Wk,
                           const float* __restrict__ Wq,
                           const float* __restrict__ Wv,
                           __bf16* __restrict__ Wt) {
    int tid = blockIdx.x * blockDim.x + threadIdx.x;
    if (tid >= 3 * kC * kH) return;
    int m = tid / (kC * kH);
    int c = (tid % (kC * kH)) / kH;
    int h = tid % kH;
    const float* W = (m == 0) ? Wk : (m == 1) ? Wq : Wv;
    Wt[(size_t)m * kC * kH + (size_t)h * kC + c] = (__bf16)W[(size_t)c * kH + h];
}

// ---------------------------------------------------------------------------
// Kernel 1: fused QKV projection.  8 waves/block, each owns a 16-row tile;
// the block shares a double-buffered LDS copy of the 3x64x32 weight tile
// (async global->LDS).  A fragments are software-pipelined one K-step ahead;
// all 12 B fragments are loaded into distinct registers before the 12 WMMAs.
// ---------------------------------------------------------------------------
__global__ void __launch_bounds__(256) qkv_proj(const float* __restrict__ x,
                                                const __bf16* __restrict__ Wt,
                                                __bf16* __restrict__ Kmat,
                                                __bf16* __restrict__ Qmat,
                                                __bf16* __restrict__ Vt) {
    __shared__ __bf16 Wl[2][3 * 64 * 32];     // [buf][m][h][32]  (12 KB each)

    const int tid  = threadIdx.x;
    const int lane = tid & 31;
    const int wid  = (blockIdx.x * blockDim.x + tid) >> 5;   // 0..2047
    const int b    = wid >> 7;            // batch
    const int t0   = (wid & 127) << 4;    // row tile base
    const int half = lane >> 4;
    const int l16  = lane & 15;

    // Async staging: 3 x 16B chunks per thread per K-step.
    auto issueW = [&](int k0, int buf) {
        #pragma unroll
        for (int c = 0; c < 3; ++c) {
            const int chunk = tid * 3 + c;          // 0..767
            const int row   = chunk >> 2;           // 0..191 (= m*64 + h)
            const int wb    = (chunk & 3) * 16;     // byte within 64B row
            const int m     = row >> 6;
            const int h     = row & 63;
            const unsigned l = lds_offset(&Wl[buf][0]) + (unsigned)(chunk * 16);
            const char* g = (const char*)Wt
                          + ((size_t)m * kC * kH + (size_t)h * kC + k0) * 2 + wb;
            async_load_b128(l, g);
        }
    };

    v8f acc[12];
    #pragma unroll
    for (int i = 0; i < 12; ++i) acc[i] = (v8f){};

    const float* xrow  = x + ((size_t)(b * kT + t0 + l16)) * kC;
    const int    cbase = 8 * half;

    // A fragment pipeline: registers hold the CURRENT step's raw f32 data.
    float4 a0 = *(const float4*)(xrow + cbase);
    float4 a1 = *(const float4*)(xrow + cbase + 4);
    float4 a2 = *(const float4*)(xrow + 16 + cbase);
    float4 a3 = *(const float4*)(xrow + 16 + cbase + 4);

    issueW(0, 0);
    for (int k0 = 0, step = 0; k0 < kC; k0 += 32, ++step) {
        const int buf = step & 1;
        wait_async();
        __syncthreads();
        if (k0 + 32 < kC) issueW(k0 + 32, buf ^ 1);

        // prefetch next step's A data (harmless re-read on the last step)
        const int kn = (k0 + 32 < kC) ? (k0 + 32) : k0;
        float4 n0 = *(const float4*)(xrow + kn + cbase);
        float4 n1 = *(const float4*)(xrow + kn + cbase + 4);
        float4 n2 = *(const float4*)(xrow + kn + 16 + cbase);
        float4 n3 = *(const float4*)(xrow + kn + 16 + cbase + 4);

        Frag a;
        cvt4(a.e + 0, a0); cvt4(a.e + 4, a1); cvt4(a.e + 8, a2); cvt4(a.e + 12, a3);

        // load ALL 12 B fragments first (distinct regs), then 12 WMMAs
        Frag bf[12];
        #pragma unroll
        for (int i = 0; i < 12; ++i) {
            const int m = i >> 2, j = i & 3;
            const __bf16* wrow = &Wl[buf][((m * 64) + 16 * j + l16) * 32 + 8 * half];
            bf[i].q[0] = *(const float4*)(wrow);
            bf[i].q[1] = *(const float4*)(wrow + 16);
        }
        #pragma unroll
        for (int i = 0; i < 12; ++i)
            acc[i] = wmma_bf16(a, bf[i], acc[i]);

        a0 = n0; a1 = n1; a2 = n2; a3 = n3;
        __syncthreads();   // buf will be overwritten next iteration
    }

    // --- store D tiles (C layout: row = r + 8*half, col = 16*j + l16) ---
    #pragma unroll
    for (int j = 0; j < 4; ++j) {
        #pragma unroll
        for (int r = 0; r < 8; ++r) {
            const int mrow = r + 8 * half;
            const int n    = 16 * j + l16;
            const size_t rowidx = ((size_t)(b * kT + t0 + mrow)) * kH + n;
            Kmat[rowidx] = (__bf16)acc[0 + j][r];
            Qmat[rowidx] = (__bf16)acc[4 + j][r];
            Vt[((size_t)b * kH + n) * kT + (t0 + mrow)] = (__bf16)acc[8 + j][r];
        }
    }
}

// ---------------------------------------------------------------------------
// Kernel 2: causal flash attention.  One block = 128 query rows of one batch
// (8 waves x 16 rows).  All waves cooperatively async-stage the 32-key
// K tile (32x64) and V tile (64x32) into double-buffered LDS; V fragments
// are hoisted ahead of the softmax so their LDS latency hides under VALU.
// ---------------------------------------------------------------------------
__global__ void __launch_bounds__(256) attn(const __bf16* __restrict__ Qmat,
                                            const __bf16* __restrict__ Kmat,
                                            const __bf16* __restrict__ Vt,
                                            float* __restrict__ out) {
    __shared__ __bf16 Kl[2][32 * 64];        // [buf][key][h]   4 KB each
    __shared__ __bf16 Vl[2][64 * 32];        // [buf][h][key]   4 KB each
    __shared__ __bf16 pbuf[8][16 * 32];      // per-wave P tile

    const int tid   = threadIdx.x;
    const int lane  = tid & 31;
    const int wslot = tid >> 5;
    const int half  = lane >> 4;
    const int l16   = lane & 15;
    const int b     = blockIdx.x >> 4;       // 16 row-blocks per batch
    const int rb    = blockIdx.x & 15;
    const int q0    = rb * 128 + wslot * 16;
    const int nkb_own = (q0 >> 5) + 1;       // key blocks up to own diagonal
    const int nkb_max = rb * 4 + 4;          // block-uniform trip count

    const __bf16* Kg = Kmat + (size_t)b * kT * kH;   // [T][64]
    const __bf16* Vg = Vt   + (size_t)b * kH * kT;   // [64][T]

    // staging: each thread moves one 16B chunk of K and one of V per block
    const int k_key = tid >> 3;              // K: 32 rows x 128 B
    const int k_hb  = (tid & 7) * 16;
    const int v_h   = tid >> 2;              // V: 64 rows x 64 B
    const int v_kb  = (tid & 3) * 16;

    auto issue = [&](int kb, int buf) {
        const unsigned lk = lds_offset(&Kl[buf][0]) + (unsigned)(tid * 16);
        const char* gk = (const char*)Kg + ((size_t)(kb * 32 + k_key) * kH) * 2 + k_hb;
        async_load_b128(lk, gk);
        const unsigned lv = lds_offset(&Vl[buf][0]) + (unsigned)(tid * 16);
        const char* gv = (const char*)Vg + ((size_t)v_h * kT + kb * 32) * 2 + v_kb;
        async_load_b128(lv, gv);
    };

    // Q fragments: h = 0..31 and 32..63 (loaded once, per-wave unique)
    Frag aq[2];
    const __bf16* qrow = Qmat + ((size_t)(b * kT + q0 + l16)) * kH;
    #pragma unroll
    for (int hh = 0; hh < 2; ++hh) {
        aq[hh].q[0] = *(const float4*)(qrow + 32 * hh + 16 * half);
        aq[hh].q[1] = *(const float4*)(qrow + 32 * hh + 16 * half + 8);
    }

    v8f o[4];
    #pragma unroll
    for (int j = 0; j < 4; ++j) o[j] = (v8f){};
    float rmax[8], rsum[8];
    #pragma unroll
    for (int r = 0; r < 8; ++r) { rmax[r] = kNegInf; rsum[r] = 0.f; }

    issue(0, 0);
    for (int kb = 0; kb < nkb_max; ++kb) {
        const int buf = kb & 1;
        wait_async();
        __syncthreads();
        if (kb + 1 < nkb_max) issue(kb + 1, buf ^ 1);

        if (kb < nkb_own) {
            const int kbase = kb * 32;

            // ---- load all 4 K fragments, then run the 4 S WMMAs ----
            Frag bk[2][2];
            #pragma unroll
            for (int t = 0; t < 2; ++t) {
                const __bf16* krow = &Kl[buf][(16 * t + l16) * 64 + 16 * half];
                bk[t][0].q[0] = *(const float4*)(krow);
                bk[t][0].q[1] = *(const float4*)(krow + 8);
                bk[t][1].q[0] = *(const float4*)(krow + 32);
                bk[t][1].q[1] = *(const float4*)(krow + 32 + 8);
            }
            v8f s[2];
            #pragma unroll
            for (int t = 0; t < 2; ++t) {
                v8f c = (v8f){};
                c = wmma_bf16(aq[0], bk[t][0], c);
                c = wmma_bf16(aq[1], bk[t][1], c);
                s[t] = c;
            }

            // ---- V fragments hoisted: their ds latency hides under softmax ----
            Frag bv[4];
            #pragma unroll
            for (int j = 0; j < 4; ++j) {
                const __bf16* vrow = &Vl[buf][(16 * j + l16) * 32 + 16 * half];
                bv[j].q[0] = *(const float4*)(vrow);
                bv[j].q[1] = *(const float4*)(vrow + 8);
            }

            // ---- scale + causal mask (diagonal block only) ----
            const bool diag = (kb == nkb_own - 1);
            #pragma unroll
            for (int t = 0; t < 2; ++t)
                #pragma unroll
                for (int r = 0; r < 8; ++r) {
                    float v = s[t][r] * kScale;
                    if (diag) {
                        const int ki = kbase + 16 * t + l16;
                        const int qi = q0 + r + 8 * half;
                        if (ki > qi) v = kNegInf;
                    }
                    s[t][r] = v;
                }

            // ---- online softmax per row (xor masks 1..8 keep halves apart) ----
            float alpha[8];
            #pragma unroll
            for (int r = 0; r < 8; ++r) {
                float m = fmaxf(s[0][r], s[1][r]);
                #pragma unroll
                for (int off = 8; off >= 1; off >>= 1)
                    m = fmaxf(m, __shfl_xor(m, off, 32));
                const float nm = fmaxf(rmax[r], m);
                alpha[r] = __expf(rmax[r] - nm);
                rmax[r] = nm;
                const float p0 = __expf(s[0][r] - nm);
                const float p1 = __expf(s[1][r] - nm);
                s[0][r] = p0; s[1][r] = p1;
                float ls = p0 + p1;
                #pragma unroll
                for (int off = 8; off >= 1; off >>= 1)
                    ls += __shfl_xor(ls, off, 32);
                rsum[r] = rsum[r] * alpha[r] + ls;
            }
            #pragma unroll
            for (int j = 0; j < 4; ++j)
                #pragma unroll
                for (int r = 0; r < 8; ++r) o[j][r] *= alpha[r];

            // ---- P (C layout) -> LDS -> A-layout fragment (wave-private) ----
            __bf16* pb = pbuf[wslot];
            #pragma unroll
            for (int t = 0; t < 2; ++t)
                #pragma unroll
                for (int r = 0; r < 8; ++r)
                    pb[(r + 8 * half) * 32 + 16 * t + l16] = (__bf16)s[t][r];
            Frag ap;
            ap.q[0] = *(const float4*)(&pb[l16 * 32 + 16 * half]);
            ap.q[1] = *(const float4*)(&pb[l16 * 32 + 16 * half + 8]);

            #pragma unroll
            for (int j = 0; j < 4; ++j)
                o[j] = wmma_bf16(ap, bv[j], o[j]);
        }
    }

    // ---- normalize and store f32 output [B][T][64] ----
    #pragma unroll
    for (int r = 0; r < 8; ++r) rsum[r] = 1.0f / rsum[r];
    #pragma unroll
    for (int j = 0; j < 4; ++j)
        #pragma unroll
        for (int r = 0; r < 8; ++r) {
            const int mrow = r + 8 * half;
            out[((size_t)(b * kT + q0 + mrow)) * kH + 16 * j + l16] = o[j][r] * rsum[r];
        }
}

// ---------------------------------------------------------------------------
extern "C" void kernel_launch(void* const* d_in, const int* in_sizes, int n_in,
                              void* d_out, int out_size, void* d_ws, size_t ws_size,
                              hipStream_t stream) {
    (void)in_sizes; (void)n_in; (void)out_size; (void)ws_size;
    const float* x  = (const float*)d_in[0];
    const float* Wk = (const float*)d_in[1];
    const float* Wq = (const float*)d_in[2];
    const float* Wv = (const float*)d_in[3];
    float* out = (float*)d_out;

    char* ws = (char*)d_ws;
    __bf16* Wt   = (__bf16*)ws;             // [3][64][768] bf16
    __bf16* Kmat = (__bf16*)(ws + kOffK);   // [B][T][64]  bf16
    __bf16* Qmat = (__bf16*)(ws + kOffQ);   // [B][T][64]  bf16
    __bf16* Vt   = (__bf16*)(ws + kOffVt);  // [B][64][T]  bf16

    wt_convert<<<(3 * kC * kH + 255) / 256, 256, 0, stream>>>(Wk, Wq, Wv, Wt);
    qkv_proj<<<256, 256, 0, stream>>>(x, Wt, Kmat, Qmat, Vt);   // 2048 waves
    attn<<<256, 256, 0, stream>>>(Qmat, Kmat, Vt, out);         // 2048 waves
}
